// StereoAttentionModule_58755152609401
// MI455X (gfx1250) — matrix operand order repaired and checked by reading, not compiled, not run on hardware
//
#include <hip/hip_runtime.h>
#include <hip/hip_bf16.h>

// ---------------- constants ----------------
constexpr int B_ = 2, C_ = 192, H_ = 48, W_ = 512;
constexpr int N_ = B_ * H_;            // 96 sequences
constexpr int EMB_ = 64, HEADS_ = 2, DOUT_ = 192, DV_ = DOUT_ / HEADS_; // 96
constexpr int MAXM_ = 64;
constexpr float THR_ = 0.01f, EPS_ = 1e-5f;
constexpr int PADL_ = 8, WP_ = W_ + 2 * PADL_;  // padded width 528

constexpr size_t SZ_X  = (size_t)N_ * C_ * WP_;            // f16 elems per transposed buffer
constexpr size_t SZ_WQ = (size_t)3 * 128 * C_;             // 73728
constexpr size_t SZ_WV = (size_t)3 * DOUT_ * C_;           // 110592
constexpr size_t SZ_WO = (size_t)DOUT_ * DOUT_;            // 36864
constexpr size_t SZ_Q  = (size_t)N_ * HEADS_ * W_ * EMB_;  // 6291456
constexpr size_t SZ_V  = (size_t)N_ * HEADS_ * W_ * DV_;   // 9437184
constexpr size_t SZ_O  = (size_t)N_ * W_ * DOUT_;          // 9437184
constexpr size_t SZ_LR = (size_t)B_ * C_ * H_ * W_;        // 9437184 (one l/r output)
constexpr size_t SZ_A  = (size_t)N_ * HEADS_ * W_ * W_;    // 50331648 (one A output)

typedef __attribute__((ext_vector_type(16))) _Float16 v16h;
typedef __attribute__((ext_vector_type(8)))  _Float16 v8h;
typedef __attribute__((ext_vector_type(8)))  float    v8f;
typedef __attribute__((ext_vector_type(4)))  int      v4i_;

// ---------------- async LDS copy (CDNA5 GLOBAL_LOAD_ASYNC_TO_LDS_B128) ----------------
#if __has_builtin(__builtin_amdgcn_global_load_async_to_lds_b128) && \
    __has_builtin(__builtin_amdgcn_s_wait_asynccnt)
#define HAVE_ASYNC_LDS 1
#else
#define HAVE_ASYNC_LDS 0
#endif

__device__ __forceinline__ void copy16B_to_lds(const _Float16* g, _Float16* s) {
#if HAVE_ASYNC_LDS
  __builtin_amdgcn_global_load_async_to_lds_b128((v4i_*)g, (v4i_*)s, 0, 0);
#else
  *(v8h*)s = *(const v8h*)g;
#endif
}

template <int N_WAIT>
__device__ __forceinline__ void wait_async() {
#if HAVE_ASYNC_LDS
  __builtin_amdgcn_s_wait_asynccnt(N_WAIT);
#endif
}

// ---------------- WMMA helpers (16x16x32 f16 -> f32) ----------------
__device__ __forceinline__ v8f wmma32(v16h a, v16h b, v8f c) {
  return __builtin_amdgcn_wmma_f32_16x16x32_f16(false, a, false, b, (short)0, c,
                                                false, false);
}

__device__ __forceinline__ v16h cat16(v8h lo, v8h hi) {
  return __builtin_shufflevector(lo, hi, 0, 1, 2, 3, 4, 5, 6, 7,
                                 8, 9, 10, 11, 12, 13, 14, 15);
}

// A fragment: 16(M) x 32(K), row-major, lda multiple of 8, base 16B aligned.
__device__ __forceinline__ v16h load_a16(const _Float16* base, int lda) {
  const int l = threadIdx.x & 31;
  const int m = l & 15, hlf = l >> 4;
  const _Float16* p = base + m * lda + hlf * 16;
  return cat16(*(const v8h*)p, *(const v8h*)(p + 8));
}

// B fragment where the source is row-major [N][K] (i.e. B = src^T, K contiguous).
__device__ __forceinline__ v16h load_bt16(const _Float16* base, int ldn) {
  const int l = threadIdx.x & 31;
  const int n = l & 15, hlf = l >> 4;
  const _Float16* p = base + (size_t)n * ldn + hlf * 16;
  return cat16(*(const v8h*)p, *(const v8h*)(p + 8));
}

// ---------------- kernel 1: layernorm + transpose + f16 pack (coalesced) ----------------
__global__ __launch_bounds__(256) void ln_pack_kernel(
    const float* __restrict__ l, const float* __restrict__ r,
    const float* __restrict__ gl, const float* __restrict__ bl,
    const float* __restrict__ gr, const float* __restrict__ br,
    _Float16* __restrict__ lnT, _Float16* __restrict__ rnT,
    _Float16* __restrict__ lsT, _Float16* __restrict__ rsT) {
  const int tx = threadIdx.x;            // 32 w lanes
  const int ty = threadIdx.y;            // 8 c groups
  const int n = blockIdx.y, b = n / H_, h = n % H_;
  const int w = blockIdx.x * 32 + tx;
  const size_t cstr = (size_t)H_ * W_;
  const size_t base = (size_t)b * C_ * cstr + (size_t)h * W_ + w;

  float sl = 0.f, sl2 = 0.f, sr = 0.f, sr2 = 0.f;
  for (int c = ty; c < C_; c += 8) {
    float lv = l[base + (size_t)c * cstr];
    float rv = r[base + (size_t)c * cstr];
    sl += lv; sl2 += lv * lv; sr += rv; sr2 += rv * rv;
  }
  __shared__ float red[4][8][33];
  red[0][ty][tx] = sl; red[1][ty][tx] = sl2; red[2][ty][tx] = sr; red[3][ty][tx] = sr2;
  __syncthreads();
  __shared__ float stat[4][33];
  if (ty == 0) {
    float a0 = 0.f, a1 = 0.f, a2 = 0.f, a3 = 0.f;
#pragma unroll
    for (int k = 0; k < 8; ++k) {
      a0 += red[0][k][tx]; a1 += red[1][k][tx];
      a2 += red[2][k][tx]; a3 += red[3][k][tx];
    }
    float ml = a0 / C_, vl = a1 / C_ - ml * ml;
    float mr = a2 / C_, vr = a3 / C_ - mr * mr;
    stat[0][tx] = ml; stat[1][tx] = rsqrtf(vl + EPS_);
    stat[2][tx] = mr; stat[3][tx] = rsqrtf(vr + EPS_);
  }
  __syncthreads();
  const float ml = stat[0][tx], rl = stat[1][tx];
  const float mr = stat[2][tx], rr = stat[3][tx];
  for (int c = ty; c < C_; c += 8) {
    float lv = l[base + (size_t)c * cstr];
    float rv = r[base + (size_t)c * cstr];
    const size_t o = ((size_t)n * C_ + c) * WP_ + PADL_ + w;
    lnT[o] = (_Float16)((lv - ml) * rl * gl[c] + bl[c]);
    rnT[o] = (_Float16)((rv - mr) * rr * gr[c] + br[c]);
    lsT[o] = (_Float16)lv;
    rsT[o] = (_Float16)rv;
  }
}

// zero the pad columns of all four transposed buffers
__global__ void pad_zero_kernel(_Float16* __restrict__ lnT, _Float16* __restrict__ rnT,
                                _Float16* __restrict__ lsT, _Float16* __restrict__ rsT) {
  int i = blockIdx.x * blockDim.x + threadIdx.x;
  if (i >= N_ * C_ * 16) return;
  int col = i % 16, nc = i / 16;
  int wcol = col < PADL_ ? col : (WP_ - 16 + col);
  size_t idx = (size_t)nc * WP_ + wcol;
  lnT[idx] = (_Float16)0.f; rnT[idx] = (_Float16)0.f;
  lsT[idx] = (_Float16)0.f; rsT[idx] = (_Float16)0.f;
}

// ---------------- weight repack ----------------
// conv weight fp32 [OC][C][3] -> f16 [3][OC][C]
__global__ void conv_w_kernel(const float* __restrict__ src, _Float16* __restrict__ dst, int OC) {
  int i = blockIdx.x * blockDim.x + threadIdx.x;
  if (i >= OC * C_ * 3) return;
  int oc = i / (C_ * 3), rem = i % (C_ * 3);
  int c = rem / 3, t = rem % 3;
  dst[((size_t)t * OC + oc) * C_ + c] = (_Float16)src[i];
}
__global__ void mat_w_kernel(const float* __restrict__ src, _Float16* __restrict__ dst, int n) {
  int i = blockIdx.x * blockDim.x + threadIdx.x;
  if (i < n) dst[i] = (_Float16)src[i];
}

// ---------------- kernel 3: conv1d-as-GEMM for Q/K/V (4 waves / block) ----------------
// X: padded f16 [N][C][WP], Wt: f16 [3][OC][C], bias fp32 [OC]
// mode 0: Q/K -> [n][h][w][d] (v8h store) ; mode 1: V -> [n][h][dv][w] (transposed)
__global__ __launch_bounds__(128) void qkv_kernel(
    const _Float16* __restrict__ X, const _Float16* __restrict__ Wt,
    const float* __restrict__ bias, _Float16* __restrict__ dst, int OC, int mode) {
  const int lane = threadIdx.x & 31;
  const int wave = threadIdx.x >> 5;
  const int hlf = lane >> 4, ncol = lane & 15;
  const int w0 = blockIdx.x * 16;
  const int oc0 = blockIdx.y * 64 + wave * 16;
  const int n = blockIdx.z;

  // X tile transposed: Xs2[j][c], j = w0-1 .. w0+16 (18 cols), stride 200 (16B aligned)
  __shared__ alignas(16) _Float16 Xs2[18 * 200];
  for (int i = threadIdx.x; i < 18 * C_; i += 128) {
    int j = i / C_, c = i % C_;
    Xs2[j * 200 + c] = X[((size_t)n * C_ + c) * WP_ + (PADL_ + w0 - 1) + j];
  }
  __syncthreads();

  v8f acc = {};
#pragma unroll
  for (int v = 0; v < 8; ++v) acc[v] = bias[oc0 + hlf * 8 + v];

  for (int t = 0; t < 3; ++t) {
    const _Float16* wb = Wt + ((size_t)t * OC + oc0) * C_;
    __builtin_prefetch(wb + (size_t)OC * C_, 0, 0);
#pragma unroll
    for (int kc = 0; kc < 6; ++kc) {
      v16h a = load_a16(wb + kc * 32, C_);
      v16h b = load_bt16(Xs2 + t * 200 + kc * 32, 200);
      acc = wmma32(a, b, acc);
    }
  }

  const int w = w0 + ncol;
  if (mode == 0) {
    const int oc = oc0 + hlf * 8;
    const int hh = oc >> 6, d0 = oc & 63;
    v8h hv;
#pragma unroll
    for (int v = 0; v < 8; ++v) hv[v] = (_Float16)acc[v];
    *(v8h*)(dst + (((size_t)n * HEADS_ + hh) * W_ + w) * EMB_ + d0) = hv;
  } else {
#pragma unroll
    for (int v = 0; v < 8; ++v) {
      const int oc = oc0 + hlf * 8 + v;
      const int hh = oc / DV_, d = oc % DV_;
      dst[(((size_t)n * HEADS_ + hh) * DV_ + d) * W_ + w] = (_Float16)acc[v];
    }
  }
}

// ---------------- zero fill for the dense A outputs ----------------
__global__ void zero_kernel(float4* __restrict__ p, size_t n4) {
  size_t i = (size_t)blockIdx.x * blockDim.x + threadIdx.x;
  if (i < n4) p[i] = make_float4(0.f, 0.f, 0.f, 0.f);
}

// ---------------- kernel 4: banded attention ----------------
// Qb,Kb: [n][h][w][d] f16 ; Vt: [n][h][dv][w] f16
__global__ __launch_bounds__(32) void attn_kernel(
    const _Float16* __restrict__ Qb, const _Float16* __restrict__ Kb,
    const _Float16* __restrict__ Vt, float* __restrict__ Aout,
    _Float16* __restrict__ Ob, int dir) {
  const int lane = threadIdx.x & 31;
  const int hlf = lane >> 4, ncol = lane & 15;
  const int q0 = blockIdx.x * 16;
  const int h = blockIdx.y;
  const int n = blockIdx.z;
  const size_t nh = (size_t)n * HEADS_ + h;
  const int ktb = (dir == 0) ? q0 : q0 - MAXM_;  // band start (multiple of 16)

  __shared__ float Sf[16 * 84];                   // 16 rows x 80 band cols
  __shared__ alignas(16) _Float16 Ah[16 * 104];   // probs, zero-padded to 96

  // ---- S = (Q K^T) / 8 over 5 key tiles ----
  v16h qa[2];
#pragma unroll
  for (int kc = 0; kc < 2; ++kc)
    qa[kc] = load_a16(Qb + (nh * W_ + q0) * EMB_ + kc * 32, EMB_);

  for (int ti = 0; ti < 5; ++ti) {
    int kk = ktb + ti * 16 + ncol;
    kk = kk < 0 ? 0 : (kk >= W_ ? W_ - 1 : kk);   // clamped; masked in softmax
    v8f acc = {};
#pragma unroll
    for (int kc = 0; kc < 2; ++kc) {
      const _Float16* p = Kb + (nh * W_ + kk) * EMB_ + kc * 32 + hlf * 16;
      v16h b = cat16(*(const v8h*)p, *(const v8h*)(p + 8));
      acc = wmma32(qa[kc], b, acc);
    }
#pragma unroll
    for (int v = 0; v < 8; ++v)
      Sf[(hlf * 8 + v) * 84 + ti * 16 + ncol] = acc[v] * 0.125f;  // 1/sqrt(64)
  }
  __syncthreads();

  // ---- banded softmax + threshold, one query row per lane 0..15 ----
  if (lane < 16) {
    const int i = q0 + lane;
    float mx = -3.4e38f;
    for (int bi = 0; bi < 80; ++bi) {
      int j = ktb + bi;
      bool valid = (bi >= lane) && (bi <= lane + MAXM_) && (j >= 0) && (j < W_);
      if (valid) { float s = Sf[lane * 84 + bi]; mx = s > mx ? s : mx; }
    }
    float sum = 0.f;
    for (int bi = 0; bi < 80; ++bi) {
      int j = ktb + bi;
      bool valid = (bi >= lane) && (bi <= lane + MAXM_) && (j >= 0) && (j < W_);
      float e = valid ? __expf(Sf[lane * 84 + bi] - mx) : 0.f;
      Sf[lane * 84 + bi] = e;
      sum += e;
    }
    const float inv = 1.f / sum;
    for (int bi = 0; bi < 96; ++bi) {
      float a = 0.f;
      if (bi < 80) {
        int j = ktb + bi;
        bool valid = (bi >= lane) && (bi <= lane + MAXM_) && (j >= 0) && (j < W_);
        a = Sf[lane * 84 + bi] * inv;
        a = (valid && a > THR_) ? a : 0.f;
        if (valid) Aout[(nh * W_ + i) * W_ + j] = a;  // rest pre-zeroed
      }
      Ah[lane * 104 + bi] = (_Float16)a;
    }
  }
  __syncthreads();

  // ---- O = Ah (16x96, zero padded) x V(96 x DV) ----
  v16h aa[3];
#pragma unroll
  for (int kc = 0; kc < 3; ++kc) aa[kc] = load_a16(Ah + kc * 32, 104);

  for (int nt = 0; nt < 6; ++nt) {
    v8f acc = {};
#pragma unroll
    for (int kc = 0; kc < 3; ++kc) {
      const int run = ktb + kc * 32 + hlf * 16;  // 16-aligned: all-in or all-out
      v16h b = {};
      if (run >= 0 && run < W_) {
        const _Float16* p = Vt + (nh * DV_ + nt * 16 + ncol) * W_ + run;
        b = cat16(*(const v8h*)p, *(const v8h*)(p + 8));
      }
      acc = wmma32(aa[kc], b, acc);
    }
#pragma unroll
    for (int v = 0; v < 8; ++v) {
      const int row = hlf * 8 + v;
      Ob[((size_t)n * W_ + q0 + row) * DOUT_ + h * DV_ + nt * 16 + ncol] =
          (_Float16)acc[v];
    }
  }
}

// ---------------- kernel 5: out-proj + residual + FF + residual -> NCHW ----------------
// Weight slices (16x192 f16 = 6KB) are double-buffered in LDS via CDNA5 async
// global->LDS copies (ASYNCcnt), overlapping the next slice's DMA with WMMAs.
__global__ __launch_bounds__(32) void proj_ff_kernel(
    const _Float16* __restrict__ Ob, const _Float16* __restrict__ Wo,
    const float* __restrict__ bo, const _Float16* __restrict__ Wf,
    const float* __restrict__ bf, const float* __restrict__ xin,
    float* __restrict__ out) {
  const int lane = threadIdx.x & 31;
  const int hlf = lane >> 4, ncol = lane & 15;
  const int w0 = blockIdx.x * 16;
  const int n = blockIdx.y;
  const int b = n / H_, hh = n % H_;

  __shared__ float Lo[16 * 196];                  // post-residual fp32
  __shared__ alignas(16) _Float16 Loh[16 * 200];  // post-residual f16
  __shared__ alignas(16) _Float16 Wst[2][16 * DOUT_];  // weight slice double buffer

  // stage one 16x192 weight slice: 384 x 16B chunks, 12 per lane
  auto stage = [&](const _Float16* src, int slot) {
#pragma unroll
    for (int it = 0; it < 12; ++it) {
      const int off = (lane + it * 32) * 8;
      copy16B_to_lds(src + off, &Wst[slot][0] + off);
    }
  };

  // hoisted A fragments of Ob (invariant over output tiles)
  v16h oa[6];
#pragma unroll
  for (int kc = 0; kc < 6; ++kc)
    oa[kc] = load_a16(Ob + ((size_t)n * W_ + w0) * DOUT_ + kc * 32, DOUT_);

  stage(Wo, 0);  // slice 0 -> slot 0

  // GEMM1: P = Ob(16x192) @ Wo^T + bo ; lo = x + P
  for (int nt = 0; nt < 12; ++nt) {
    // prefetch next slice (slice 12 == first slice of Wf)
    const _Float16* nsrc = (nt < 11) ? Wo + (size_t)(nt + 1) * 16 * DOUT_ : Wf;
    stage(nsrc, (nt + 1) & 1);
    wait_async<12>();  // current slice (slot nt&1) has landed
    const _Float16* wsl = &Wst[nt & 1][0];

    const int o0 = nt * 16;
    v8f acc = {};
#pragma unroll
    for (int v = 0; v < 8; ++v) acc[v] = bo[o0 + ncol];
#pragma unroll
    for (int kc = 0; kc < 6; ++kc) {
      v16h bb = load_bt16(wsl + kc * 32, DOUT_);
      acc = wmma32(oa[kc], bb, acc);
    }
    const int o = o0 + ncol;
    const float* xrow = xin + (((size_t)b * C_ + o) * H_ + hh) * W_ + w0 + hlf * 8;
    float4 x0 = *(const float4*)xrow;
    float4 x1 = *(const float4*)(xrow + 4);
    const float xv[8] = {x0.x, x0.y, x0.z, x0.w, x1.x, x1.y, x1.z, x1.w};
#pragma unroll
    for (int v = 0; v < 8; ++v) {
      const int row = hlf * 8 + v;
      const float lv = xv[v] + acc[v];
      Lo[row * 196 + o] = lv;
      Loh[row * 200 + o] = (_Float16)lv;
    }
  }
  __syncthreads();

  // hoisted A fragments of lo
  v16h la[6];
#pragma unroll
  for (int kc = 0; kc < 6; ++kc) la[kc] = load_a16(Loh + kc * 32, 200);

  // GEMM2: out = lo + (lo @ Wf^T + bf), scattered to NCHW (8 contiguous w per lane)
  for (int nt = 0; nt < 12; ++nt) {
    if (nt < 11) {
      stage(Wf + (size_t)(nt + 1) * 16 * DOUT_, (nt + 1) & 1);
      wait_async<12>();
    } else {
      wait_async<0>();
    }
    const _Float16* wsl = &Wst[nt & 1][0];

    const int o0 = nt * 16;
    v8f acc = {};
#pragma unroll
    for (int v = 0; v < 8; ++v) acc[v] = bf[o0 + ncol];
#pragma unroll
    for (int kc = 0; kc < 6; ++kc) {
      v16h bb = load_bt16(wsl + kc * 32, DOUT_);
      acc = wmma32(la[kc], bb, acc);
    }
    const int o = o0 + ncol;
    float res[8];
#pragma unroll
    for (int v = 0; v < 8; ++v) {
      const int row = hlf * 8 + v;
      res[v] = Lo[row * 196 + o] + acc[v];
    }
    float* orow = out + (((size_t)b * C_ + o) * H_ + hh) * W_ + w0 + hlf * 8;
    *(float4*)orow       = make_float4(res[0], res[1], res[2], res[3]);
    *(float4*)(orow + 4) = make_float4(res[4], res[5], res[6], res[7]);
  }
}

// ---------------- launch ----------------
extern "C" void kernel_launch(void* const* d_in, const int* in_sizes, int n_in,
                              void* d_out, int out_size, void* d_ws, size_t ws_size,
                              hipStream_t stream) {
  (void)in_sizes; (void)n_in; (void)out_size; (void)ws_size;
  const float* l = (const float*)d_in[0];
  const float* r = (const float*)d_in[1];
  const float* gl = (const float*)d_in[2];
  const float* bl = (const float*)d_in[3];
  const float* gr = (const float*)d_in[4];
  const float* br = (const float*)d_in[5];
  const float* wq[2] = {(const float*)d_in[6],  (const float*)d_in[14]};
  const float* bq[2] = {(const float*)d_in[7],  (const float*)d_in[15]};
  const float* wk[2] = {(const float*)d_in[8],  (const float*)d_in[16]};
  const float* bk[2] = {(const float*)d_in[9],  (const float*)d_in[17]};
  const float* wv[2] = {(const float*)d_in[10], (const float*)d_in[18]};
  const float* bv[2] = {(const float*)d_in[11], (const float*)d_in[19]};
  const float* wo[2] = {(const float*)d_in[12], (const float*)d_in[20]};
  const float* bo[2] = {(const float*)d_in[13], (const float*)d_in[21]};
  const float* fw[2] = {(const float*)d_in[22], (const float*)d_in[24]};
  const float* fb[2] = {(const float*)d_in[23], (const float*)d_in[25]};

  _Float16* p = (_Float16*)d_ws;
  _Float16* lnT = p; p += SZ_X;
  _Float16* rnT = p; p += SZ_X;
  _Float16* lsT = p; p += SZ_X;
  _Float16* rsT = p; p += SZ_X;
  _Float16 *wqf[2], *wkf[2], *wvf[2], *wof[2], *fff[2], *qb[2], *kb[2], *vb[2], *ob[2];
  for (int i = 0; i < 2; ++i) { wqf[i] = p; p += SZ_WQ; wkf[i] = p; p += SZ_WQ; wvf[i] = p; p += SZ_WV; }
  for (int i = 0; i < 2; ++i) { wof[i] = p; p += SZ_WO; fff[i] = p; p += SZ_WO; }
  for (int i = 0; i < 2; ++i) { qb[i] = p; p += SZ_Q; kb[i] = p; p += SZ_Q; vb[i] = p; p += SZ_V; }
  for (int i = 0; i < 2; ++i) { ob[i] = p; p += SZ_O; }

  float* out  = (float*)d_out;
  float* lout = out;
  float* rout = out + SZ_LR;
  float* Artl = out + 2 * SZ_LR;
  float* Altr = Artl + SZ_A;

  ln_pack_kernel<<<dim3(W_ / 32, N_), dim3(32, 8), 0, stream>>>(
      l, r, gl, bl, gr, br, lnT, rnT, lsT, rsT);
  pad_zero_kernel<<<(N_ * C_ * 16 + 255) / 256, 256, 0, stream>>>(lnT, rnT, lsT, rsT);

  for (int ps = 0; ps < 2; ++ps) {
    conv_w_kernel<<<288, 256, 0, stream>>>(wq[ps], wqf[ps], 128);
    conv_w_kernel<<<288, 256, 0, stream>>>(wk[ps], wkf[ps], 128);
    conv_w_kernel<<<432, 256, 0, stream>>>(wv[ps], wvf[ps], 192);
    mat_w_kernel<<<144, 256, 0, stream>>>(wo[ps], wof[ps], (int)SZ_WO);
    mat_w_kernel<<<144, 256, 0, stream>>>(fw[ps], fff[ps], (int)SZ_WO);
  }

  const _Float16* qsrc[2] = {lnT, rnT};  // rtl: Q from ln_, K from rn_, V from rs
  const _Float16* ksrc[2] = {rnT, lnT};  // ltr: Q from rn_, K from ln_, V from ls
  const _Float16* vsrc[2] = {rsT, lsT};
  for (int ps = 0; ps < 2; ++ps) {
    qkv_kernel<<<dim3(W_ / 16, 128 / 64, N_), 128, 0, stream>>>(qsrc[ps], wqf[ps], bq[ps], qb[ps], 128, 0);
    qkv_kernel<<<dim3(W_ / 16, 128 / 64, N_), 128, 0, stream>>>(ksrc[ps], wkf[ps], bk[ps], kb[ps], 128, 0);
    qkv_kernel<<<dim3(W_ / 16, 192 / 64, N_), 128, 0, stream>>>(vsrc[ps], wvf[ps], bv[ps], vb[ps], 192, 1);
  }

  const size_t n4 = (2 * SZ_A) / 4;
  zero_kernel<<<(unsigned)((n4 + 255) / 256), 256, 0, stream>>>((float4*)Artl, n4);

  attn_kernel<<<dim3(W_ / 16, HEADS_, N_), 32, 0, stream>>>(qb[0], kb[0], vb[0], Artl, ob[0], 0);
  attn_kernel<<<dim3(W_ / 16, HEADS_, N_), 32, 0, stream>>>(qb[1], kb[1], vb[1], Altr, ob[1], 1);

  proj_ff_kernel<<<dim3(W_ / 16, N_), 32, 0, stream>>>(ob[0], wof[0], bo[0], fff[0], fb[0], l, lout);
  proj_ff_kernel<<<dim3(W_ / 16, N_), 32, 0, stream>>>(ob[1], wof[1], bo[1], fff[1], fb[1], r, rout);
}
